// MultiHeadAttention_14688788152931
// MI455X (gfx1250) — matrix operand order repaired and checked by reading, not compiled, not run on hardware
//
#include <hip/hip_runtime.h>

// ---------------------------------------------------------------------------
// MI455X (gfx1250, wave32) multi-head attention:
//   Q/K/V projections (f16 WMMA, fp32 accum) -> flash attention (online
//   softmax, f16 WMMA) -> output projection (f16 WMMA, fp32 out).
// B=4, S=2048, D=1024, H=16, dh=64.
//
// CDNA5 specifics used:
//  - v_wmma_f32_16x16x32_f16 (wave32 WMMA, fp32 accumulate)
//  - global_load_async_to_lds_b128 (+ s_wait_asynccnt) for tile staging
//  - LDS tiles stored pre-swizzled in WMMA fragment order -> each fragment
//    is two ds_load_b128 per lane, XOR-swizzled for minimal bank conflicts.
// ---------------------------------------------------------------------------

typedef unsigned int u32;
typedef unsigned short u16;
typedef _Float16 f16;
typedef __attribute__((ext_vector_type(16))) _Float16 v16h;
typedef __attribute__((ext_vector_type(8)))  float    v8f;
typedef __attribute__((ext_vector_type(4)))  u32      u32x4;

union FragH { u32 u[8]; u32x4 q4[2]; v16h v; };   // 16 halves == 8 dwords

__device__ inline v8f zero8() { v8f z; for (int i = 0; i < 8; ++i) z[i] = 0.f; return z; }

#define WMMA_F16(a, b, c) \
  __builtin_amdgcn_wmma_f32_16x16x32_f16(false, (a), false, (b), (short)0, (c), false, false)

// Async 16B copy global -> LDS (GVS mode: 64-bit SGPR base + 32-bit lane offset).
__device__ inline void async_b128(u32 lds_byte, u32 voff_byte, const void* sbase) {
  asm volatile("global_load_async_to_lds_b128 %0, %1, %2"
               :: "v"(lds_byte), "v"(voff_byte), "s"(sbase) : "memory");
}
__device__ inline void wait_async0() {
  asm volatile("s_wait_asynccnt 0x0" ::: "memory");
}

// Fragment-order LDS layout: frag f occupies dwords [f*256, f*256+256);
// lane L owns 8 dwords at L*8, with the upper/lower quad XOR-swapped by
// bit3 of the lane to kill bank conflicts on b128 reads.
__device__ inline void ld_frag(const u32* lds, int fragBase, int lane, FragH& f) {
  const int xo = ((lane >> 3) & 1) << 2;
  f.q4[0] = *(const u32x4*)(lds + fragBase + lane * 8 + xo);
  f.q4[1] = *(const u32x4*)(lds + fragBase + lane * 8 + (4 ^ xo));
}

// ---------------------------------------------------------------------------
// fp32 -> f16 conversion (grid-stride)
// ---------------------------------------------------------------------------
__global__ __launch_bounds__(256) void cvt_f32_f16(const float* __restrict__ in,
                                                   f16* __restrict__ out, int n) {
  int i = blockIdx.x * blockDim.x + threadIdx.x;
  int stride = gridDim.x * blockDim.x;
  for (; i < n; i += stride) out[i] = (f16)in[i];
}

// ---------------------------------------------------------------------------
// C[M,N] = A[M,K] * B[N,K]^T   (torch Linear: y = x @ W^T, W is [out,in])
// A, B f16 (dword pointers, K/2 dwords per row). Block: 256 threads = 8 waves,
// tile 128x128; wave (wm,wn) computes rows wm*32..+31, cols wn*64..+63
// (2 x 4 WMMA tiles). Tiles staged with async b128 into fragment order.
// ---------------------------------------------------------------------------
template <typename OutT>
__global__ __launch_bounds__(256) void gemm_xwt(const u32* __restrict__ A,
                                                const u32* __restrict__ Bw,
                                                OutT* __restrict__ C,
                                                int M, int N, int Kd) {
  __shared__ u32 Al[2048];   // 8 frags (8 M-subtiles x K=32)
  __shared__ u32 Bl[2048];   // 8 frags (8 N-subtiles x K=32)
  const int kdwPitch = Kd >> 1;
  const int m0 = blockIdx.y * 128;
  const int n0 = blockIdx.x * 128;
  const int tid  = threadIdx.x;
  const int wave = tid >> 5, lane = tid & 31;
  const int g = lane >> 4, ln = lane & 15;
  const int wm = wave >> 1, wn = wave & 1;
  const u32 aBase = (u32)(size_t)Al;
  const u32 bBase = (u32)(size_t)Bl;

  v8f acc[2][4];
  for (int i = 0; i < 2; ++i)
    for (int j = 0; j < 4; ++j) acc[i][j] = zero8();

  for (int kk = 0; kk < Kd; kk += 32) {
    const int kdw = kk >> 1;
    // 512 transfers of 16B per tile; r = tile row, c0 = first of 4 dwords.
    for (int t = tid; t < 512; t += 256) {
      const int r = t >> 2, c0 = (t & 3) << 2;
      {  // A tile: lane = (m&15) + ((c>>2)&1)*16 ; v = ((c>>3)<<2)|(c&3)
        const int lane_t = (r & 15) + (((c0 >> 2) & 1) << 4);
        const int vhi = (c0 >> 3) << 2;
        const int xo = ((lane_t >> 3) & 1) << 2;
        const int dst = (r >> 4) * 256 + lane_t * 8 + (vhi ^ xo);
        async_b128(aBase + dst * 4,
                   (u32)(((m0 + r) * kdwPitch + kdw + c0) * 4), (const void*)A);
      }
      {  // B tile: lane = (n&15) + ((c>>2)&1)*16 ; v = ((c>>3)<<2)|(c&3)
        const int lane_t = (r & 15) + (((c0 >> 2) & 1) << 4);
        const int vhi = (c0 >> 3) << 2;
        const int xo = ((lane_t >> 3) & 1) << 2;
        const int dst = (r >> 4) * 256 + lane_t * 8 + (vhi ^ xo);
        async_b128(bBase + dst * 4,
                   (u32)(((n0 + r) * kdwPitch + kdw + c0) * 4), (const void*)Bw);
      }
    }
    wait_async0();
    __syncthreads();

    FragH fa0, fa1, fb;
    ld_frag(Al, (wm * 2 + 0) * 256, lane, fa0);
    ld_frag(Al, (wm * 2 + 1) * 256, lane, fa1);
    for (int ns = 0; ns < 4; ++ns) {
      ld_frag(Bl, (wn * 4 + ns) * 256, lane, fb);
      acc[0][ns] = WMMA_F16(fa0.v, fb.v, acc[0][ns]);
      acc[1][ns] = WMMA_F16(fa1.v, fb.v, acc[1][ns]);
    }
    __syncthreads();
  }

  for (int i = 0; i < 2; ++i)
    for (int ns = 0; ns < 4; ++ns) {
      const int col = n0 + wn * 64 + ns * 16 + ln;
      for (int v = 0; v < 8; ++v) {
        const int row = m0 + wm * 32 + i * 16 + v + (g << 3);
        C[(long)row * N + col] = (OutT)acc[i][ns][v];
      }
    }
}

// ---------------------------------------------------------------------------
// Flash attention: one block per (b, h, 64-query tile). 128 threads = 4 waves,
// each wave owns 16 query rows. dh = 64 (2 WMMA K-steps of 32).
// Q/K/V are f16 [B*S, 1024]; head h occupies columns h*64..h*64+63.
// K staged async into fragment order; V transposed manually into fragment
// order; P scatter-stored directly in A-fragment order.
// ---------------------------------------------------------------------------
__global__ __launch_bounds__(128) void flash_attn(const u32* __restrict__ Q,
                                                  const u32* __restrict__ Kg,
                                                  const u32* __restrict__ Vg,
                                                  const int* __restrict__ vlens,
                                                  f16* __restrict__ ctx) {
  __shared__ u32 Kl[2048];   // B-frags: frag = keySub*2 + ks
  __shared__ u32 Vl[2048];   // B-frags (transposed): frag = dhSub*2 + ks
  __shared__ u32 Pl[2048];   // A-frags: frag = wave*2 + ks (per-wave private)
  const int qt = blockIdx.x, h = blockIdx.y, b = blockIdx.z;
  const int tid  = threadIdx.x;
  const int wave = tid >> 5, lane = tid & 31;
  const int g = lane >> 4, ln = lane & 15;
  const u16* Vh = (const u16*)Vg;
  f16* Ph = (f16*)Pl;
  const u32 kBase = (u32)(size_t)Kl;

  int vl = vlens[b];
  if (vl < 1) vl = 1;
  if (vl > 2048) vl = 2048;
  const int nt = (vl + 63) >> 6;   // key tiles with >=1 valid column

  // Q A-fragments straight from global (two 16B loads per k-step)
  FragH faQ[2];
  {
    const long qrow = (long)b * 2048 + qt * 64 + wave * 16 + ln;
    const u32* qr = Q + qrow * 512 + h * 32;
    for (int ks = 0; ks < 2; ++ks) {
      faQ[ks].q4[0] = *(const u32x4*)(qr + ks * 16 + (g << 2));
      faQ[ks].q4[1] = *(const u32x4*)(qr + ks * 16 + 8 + (g << 2));
    }
  }

  float mrow[8], lrow[8];
  for (int v = 0; v < 8; ++v) { mrow[v] = -3.0e38f; lrow[v] = 0.f; }
  v8f oacc[4];
  for (int i = 0; i < 4; ++i) oacc[i] = zero8();

  for (int kt = 0; kt < nt; ++kt) {
    const long krow0 = (long)b * 2048 + kt * 64;
    // K tile: async b128, fragment order (B-frag over 64-half K dim)
    for (int t = tid; t < 512; t += 128) {
      const int r = t >> 3, c0 = (t & 7) << 2;         // r=key, c0 in 0..28
      const int lane_t = (r & 15) + (((c0 >> 2) & 1) << 4);
      const int vhi = ((c0 >> 3) & 1) << 2;
      const int xo = ((lane_t >> 3) & 1) << 2;
      const int dst = ((r >> 4) * 2 + (c0 >> 4)) * 256 + lane_t * 8 + (vhi ^ xo);
      async_b128(kBase + dst * 4,
                 (u32)(((u32)(krow0 + r) * 512 + h * 32 + c0) * 4), (const void*)Kg);
    }
    // V tile transposed [d][k] into fragment order (half-granularity gathers)
    for (int idx = tid; idx < 2048; idx += 128) {
      const int d = idx >> 5, jp = idx & 31;           // d = dh row, jp = key pair
      u32 lo = Vh[(krow0 + 2 * jp)     * 1024 + h * 64 + d];
      u32 hi = Vh[(krow0 + 2 * jp + 1) * 1024 + h * 64 + d];
      const int lane_t = (d & 15) + (((jp >> 2) & 1) << 4);
      const int v = (((jp >> 3) & 1) << 2) | (jp & 3);
      const int xo = ((lane_t >> 3) & 1) << 2;
      const int dst = ((d >> 4) * 2 + (jp >> 4)) * 256 + lane_t * 8 + ((v & 4) ^ xo) + (v & 3);
      Vl[dst] = lo | (hi << 16);
    }
    wait_async0();
    __syncthreads();

    // S = Q K^T (f32 accum)
    v8f s[4];
    for (int ns = 0; ns < 4; ++ns) {
      s[ns] = zero8();
      for (int ks = 0; ks < 2; ++ks) {
        FragH fb;
        ld_frag(Kl, (ns * 2 + ks) * 256, lane, fb);
        s[ns] = WMMA_F16(faQ[ks].v, fb.v, s[ns]);
      }
    }

    // scale (1/sqrt(64)), mask, per-row max (reduce over 16 N lanes)
    float rm[8];
    for (int v = 0; v < 8; ++v) rm[v] = -3.0e38f;
    for (int ns = 0; ns < 4; ++ns) {
      const bool ok = (kt * 64 + ns * 16 + ln) < vl;
      for (int v = 0; v < 8; ++v) {
        float x = ok ? s[ns][v] * 0.125f : -1.0e9f;
        s[ns][v] = x;
        rm[v] = fmaxf(rm[v], x);
      }
    }
    for (int off = 1; off < 16; off <<= 1)
      for (int v = 0; v < 8; ++v)
        rm[v] = fmaxf(rm[v], __shfl_xor(rm[v], off, 32));

    float alpha[8];
    for (int v = 0; v < 8; ++v) {
      const float mn = fmaxf(mrow[v], rm[v]);
      alpha[v] = __expf(mrow[v] - mn);
      mrow[v] = mn;
    }

    // P = exp(S - m); row sums; scatter P into A-fragment order (own band)
    float rs[8];
    for (int v = 0; v < 8; ++v) rs[v] = 0.f;
    for (int ns = 0; ns < 4; ++ns) {
      const int ks = ns >> 1;
      const int vfrag = ((ns & 1) << 2) | ((ln >> 1) & 3);
      const int gk = (ln >> 3) & 1;      // K-half selector within frag
      const int sbit = ln & 1;
      for (int v = 0; v < 8; ++v) {
        const float p = __expf(s[ns][v] - mrow[v]);
        rs[v] += p;
        const int lane_t = v + (g << 3) + (gk << 4);   // row = v+8g, Kgroup gk
        const int xo = g << 2;                          // ((lane_t>>3)&1)<<2
        const int dst = (wave * 2 + ks) * 256 + lane_t * 8 + ((vfrag & 4) ^ xo) + (vfrag & 3);
        Ph[dst * 2 + sbit] = (f16)p;
      }
    }
    for (int off = 1; off < 16; off <<= 1)
      for (int v = 0; v < 8; ++v)
        rs[v] += __shfl_xor(rs[v], off, 32);
    for (int v = 0; v < 8; ++v)
      lrow[v] = lrow[v] * alpha[v] + rs[v];

    // rescale running O, then O += P·V
    for (int ns = 0; ns < 4; ++ns)
      for (int v = 0; v < 8; ++v)
        oacc[ns][v] *= alpha[v];

    FragH faP[2];   // own band -> no cross-wave barrier needed
    ld_frag(Pl, (wave * 2 + 0) * 256, lane, faP[0]);
    ld_frag(Pl, (wave * 2 + 1) * 256, lane, faP[1]);
    for (int ns = 0; ns < 4; ++ns) {
      for (int ks = 0; ks < 2; ++ks) {
        FragH fbv;
        ld_frag(Vl, (ns * 2 + ks) * 256, lane, fbv);
        oacc[ns] = WMMA_F16(faP[ks].v, fbv.v, oacc[ns]);
      }
    }
    __syncthreads();   // protect K/V tiles before next stage
  }

  // normalize and write context in [B,S,H*dh] layout (f16)
  for (int v = 0; v < 8; ++v) {
    const float inv = 1.0f / lrow[v];
    const long row = (long)b * 2048 + qt * 64 + wave * 16 + v + (g << 3);
    for (int ns = 0; ns < 4; ++ns)
      ctx[row * 1024 + h * 64 + ns * 16 + ln] = (f16)(oacc[ns][v] * inv);
  }
}

// ---------------------------------------------------------------------------
// Host launch
// ---------------------------------------------------------------------------
extern "C" void kernel_launch(void* const* d_in, const int* in_sizes, int n_in,
                              void* d_out, int out_size, void* d_ws, size_t ws_size,
                              hipStream_t stream) {
  (void)in_sizes; (void)n_in; (void)out_size; (void)ws_size;
  const float* q  = (const float*)d_in[0];
  const float* k  = (const float*)d_in[1];
  const float* v  = (const float*)d_in[2];
  const float* Wq = (const float*)d_in[3];
  const float* Wk = (const float*)d_in[4];
  const float* Wv = (const float*)d_in[5];
  const float* Wo = (const float*)d_in[6];
  const int* vlens = (const int*)d_in[7];

  const size_t XN = 8192ull * 1024ull;   // activation elems
  const size_t WN = 1024ull * 1024ull;   // weight elems
  f16* Xq  = (f16*)d_ws;        // workspace (f16): 3*X + 4*W + 3*X + X ~ 120 MB
  f16* Xk  = Xq  + XN;
  f16* Xv  = Xk  + XN;
  f16* Wqh = Xv  + XN;
  f16* Wkh = Wqh + WN;
  f16* Wvh = Wkh + WN;
  f16* Woh = Wvh + WN;
  f16* Qh  = Woh + WN;
  f16* Kh  = Qh  + XN;
  f16* Vh  = Kh  + XN;
  f16* Ctx = Vh  + XN;

  cvt_f32_f16<<<2048, 256, 0, stream>>>(q,  Xq,  (int)XN);
  cvt_f32_f16<<<2048, 256, 0, stream>>>(k,  Xk,  (int)XN);
  cvt_f32_f16<<<2048, 256, 0, stream>>>(v,  Xv,  (int)XN);
  cvt_f32_f16<<<512,  256, 0, stream>>>(Wq, Wqh, (int)WN);
  cvt_f32_f16<<<512,  256, 0, stream>>>(Wk, Wkh, (int)WN);
  cvt_f32_f16<<<512,  256, 0, stream>>>(Wv, Wvh, (int)WN);
  cvt_f32_f16<<<512,  256, 0, stream>>>(Wo, Woh, (int)WN);

  dim3 gg(8, 64);   // N/128 x M/128
  gemm_xwt<f16><<<gg, 256, 0, stream>>>((const u32*)Xq, (const u32*)Wqh, Qh, 8192, 1024, 1024);
  gemm_xwt<f16><<<gg, 256, 0, stream>>>((const u32*)Xk, (const u32*)Wkh, Kh, 8192, 1024, 1024);
  gemm_xwt<f16><<<gg, 256, 0, stream>>>((const u32*)Xv, (const u32*)Wvh, Vh, 8192, 1024, 1024);

  flash_attn<<<dim3(32, 16, 4), 128, 0, stream>>>((const u32*)Qh, (const u32*)Kh,
                                                  (const u32*)Vh, vlens, Ctx);

  gemm_xwt<float><<<gg, 256, 0, stream>>>((const u32*)Ctx, (const u32*)Woh,
                                          (float*)d_out, 8192, 1024, 1024);
}
